// GraphExpert_25632364822535
// MI455X (gfx1250) — matrix operand order repaired and checked by reading, not compiled, not run on hardware
//
#include <hip/hip_runtime.h>
#include <hip/hip_bf16.h>
#include <math.h>
#include <stdint.h>

typedef __bf16 bf16;
typedef __bf16 v16bf __attribute__((ext_vector_type(16)));
typedef float  v8f   __attribute__((ext_vector_type(8)));

#define S_PER_WG 8
#define N_NODES 1000
#define B_SZ 16
#define NSAMP (B_SZ * N_NODES)      // 16000
#define KPAD 7712                   // 241 * 32 (7682 padded)
#define NEDGE 16000
#define NEP (NEDGE + N_NODES)       // edges + self loops

__device__ __forceinline__ v8f wmma_bf(v16bf a, v16bf b, v8f c) {
    return __builtin_amdgcn_wmma_f32_16x16x32_bf16(false, a, false, b, (short)0, c,
                                                   false, false);
}

union Frag {
    v16bf v;
    bf16  h[16];
    uint4 q[2];
};

// load 2x8 contiguous bf16 halves from two 16B-aligned LDS locations
__device__ __forceinline__ v16bf load_frag16(const bf16* p0, const bf16* p1) {
    Frag f;
    f.q[0] = *(const uint4*)p0;
    f.q[1] = *(const uint4*)p1;
    return f.v;
}

// ---------------------------------------------------------------------------
// Tensor Data Mover: 2-D tile load global->LDS (bf16 elements).
// D# packed per CDNA5 ISA:
//   group0: count=1 | lds_addr | global_addr[56:0] | type=2
//   group1: data_size=2B, tensor_dim = tile_dim (no OOB), dim0_stride = ld
// ---------------------------------------------------------------------------
#if __has_builtin(__builtin_amdgcn_tensor_load_to_lds)
#define USE_TDM 1
#else
#define USE_TDM 0
#endif

#if USE_TDM
typedef unsigned int u32x4 __attribute__((ext_vector_type(4)));
typedef int          i32x4 __attribute__((ext_vector_type(4)));
typedef int          i32x8 __attribute__((ext_vector_type(8)));

__device__ __forceinline__ void tdm_load_tile_2d(void* lds_dst, const void* gsrc,
                                                 unsigned d0_elems, unsigned d1_rows,
                                                 unsigned long long stride_elems)
{
    unsigned long long ga = (unsigned long long)(uintptr_t)gsrc;
    u32x4 g0;
    g0[0] = 1u;                                   // count=1 (valid user D#)
    g0[1] = (unsigned)(uintptr_t)lds_dst;         // lds_addr (low 32 = LDS offset)
    g0[2] = (unsigned)ga;                         // global_addr[31:0]
    g0[3] = (unsigned)(ga >> 32) | (2u << 30);    // global_addr[56:32] | type=2
    i32x8 g1;
    g1[0] = 0x10000;                              // data_size=1 (2 bytes)
    g1[1] = (int)((d0_elems & 0xFFFFu) << 16);                     // tensor_dim0 lo
    g1[2] = (int)((d0_elems >> 16) | ((d1_rows & 0xFFFFu) << 16)); // t_dim0 hi|t_dim1 lo
    g1[3] = (int)((d1_rows >> 16) | (d0_elems << 16));             // t_dim1 hi|tile_dim0
    g1[4] = (int)(d1_rows & 0xFFFFu);                              // tile_dim1
    g1[5] = (int)(unsigned)stride_elems;                           // dim0_stride lo
    g1[6] = (int)((unsigned)(stride_elems >> 32) & 0xFFFFu);       // dim0_stride hi
    g1[7] = 0;
    i32x4 z4 = {0, 0, 0, 0};
#if __clang_major__ >= 23
    i32x8 z8 = {0, 0, 0, 0, 0, 0, 0, 0};
    __builtin_amdgcn_tensor_load_to_lds(g0, g1, z4, z4, z8, 0);
#else
    __builtin_amdgcn_tensor_load_to_lds(g0, g1, z4, z4, 0);
#endif
}

__device__ __forceinline__ void tdm_wait(int n) {
#if __has_builtin(__builtin_amdgcn_s_wait_tensorcnt)
    if (n == 0) __builtin_amdgcn_s_wait_tensorcnt((short)0);
    else        __builtin_amdgcn_s_wait_tensorcnt((short)2);
#else
    if (n == 0) asm volatile("s_wait_tensorcnt 0x0" ::: "memory");
    else        asm volatile("s_wait_tensorcnt 0x2" ::: "memory");
#endif
}
#endif  // USE_TDM

// ---------------------------------------------------------------------------
// Kernel A: fused conv1(relu) -> conv2(relu), implicit-GEMM via WMMA.
// Branchless im2col gathers: weight K-pad rows are zero and out-of-range M
// rows only feed discarded outputs, so no guards needed around LDS reads.
// ---------------------------------------------------------------------------
__global__ __launch_bounds__(256)
void conv_fused_kernel(const float* __restrict__ x,   // [16000][8][64]
                       const float* __restrict__ w1,  // [64][8][3]
                       const float* __restrict__ b1,  // [64]
                       const float* __restrict__ w2,  // [128][64][3]
                       const float* __restrict__ b2,  // [128]
                       bf16* __restrict__ h2)         // [16000][KPAD]
{
    __shared__ __align__(16) bf16 sX[S_PER_WG][8][64];    // 8 KB
    __shared__ __align__(16) bf16 sW1[64][32];            // 4 KB (K padded 24->32, pad=0)
    __shared__ __align__(16) bf16 sW2[128][192];          // 48 KB
    __shared__ __align__(16) bf16 sO1[S_PER_WG][64][64];  // 64 KB ([s][t1][c])

    const int t    = threadIdx.x;
    const int lane = t & 31;
    const int wave = t >> 5;
    const int s0   = blockIdx.x * S_PER_WG;
    const int hi8  = (lane >= 16) ? 8 : 0;
    const int khi  = (lane >= 16) ? 16 : 0;
    const int lm   = lane & 15;

    for (int i = t; i < S_PER_WG * 8 * 64; i += 256)
        ((bf16*)sX)[i] = (bf16)x[(size_t)s0 * 512 + i];
    for (int i = t; i < 64 * 32; i += 256) {
        int k = i & 31;
        ((bf16*)sW1)[i] = (k < 24) ? (bf16)w1[(i >> 5) * 24 + k] : (bf16)0.0f;
    }
    for (int i = t; i < 128 * 192; i += 256)
        ((bf16*)sW2)[i] = (bf16)w2[i];
    __syncthreads();

    // ---- conv1 as GEMM: M=(s,t1) tiles, N=64 (4 tiles), K=24->32 (1 chunk)
    for (int idx = wave; idx < S_PER_WG * 4 * 4; idx += 8) {
        int s = idx >> 4, mt = (idx >> 2) & 3, nt = idx & 3;
        int t1 = mt * 16 + lm;
        Frag fa;
#pragma unroll
        for (int h = 0; h < 16; ++h) {
            int k   = (h < 8) ? (hi8 + h) : (8 + hi8 + h);  // A lane K map
            int cin = k / 3, kk = k - cin * 3;
            fa.h[h] = sX[s][cin & 7][t1 + kk];  // pad cols hit zero weights
        }
        v16bf fb = load_frag16(&sW1[nt * 16 + lm][hi8], &sW1[nt * 16 + lm][16 + hi8]);
        v8f acc = {};
        acc = wmma_bf(fa.v, fb, acc);
        float bias = b1[nt * 16 + lm];
#pragma unroll
        for (int r = 0; r < 8; ++r) {
            int row = mt * 16 + r + hi8;   // t1 (rows 62,63 pad, never read)
            sO1[s][row][nt * 16 + lm] = (bf16)fmaxf(acc[r] + bias, 0.f);
        }
    }
    __syncthreads();

    // ---- conv2 as GEMM: M=(s,t2) tiles, N=128 (8 tiles), K=192 (6 chunks)
    for (int idx = wave; idx < S_PER_WG * 4 * 8; idx += 8) {
        int s = idx >> 5, mt = (idx >> 3) & 3, nt = idx & 7;
        int t2 = mt * 16 + lm;
        v8f acc = {};
        for (int kc = 0; kc < 6; ++kc) {
            Frag fa;
#pragma unroll
            for (int h = 0; h < 16; ++h) {
                int k = kc * 32 + ((h < 8) ? (hi8 + h) : (8 + hi8 + h));
                int c = k / 3, kk = k - c * 3;
                fa.h[h] = sO1[s][t2 + kk][c];  // OOB rows -> garbage into pad rows only
            }
            v16bf fb = load_frag16(&sW2[nt * 16 + lm][kc * 32 + hi8],
                                   &sW2[nt * 16 + lm][kc * 32 + 16 + hi8]);
            acc = wmma_bf(fa.v, fb, acc);
        }
        float bias = b2[nt * 16 + lm];
        int   c    = nt * 16 + lm;
        size_t base = (size_t)(s0 + s) * KPAD;
#pragma unroll
        for (int r = 0; r < 8; ++r) {
            int row = mt * 16 + r + hi8;  // t2
            if (row < 60)
                h2[base + (size_t)c * 60 + row] = (bf16)fmaxf(acc[r] + bias, 0.f);
        }
    }
}

// ---------------------------------------------------------------------------
// Shared GEMM: C[M x 128] = A[M x lda] * Bt^T (Bt is [128][ldb], bf16), f32 out.
// Double-buffered TDM tile loads (fallback: cooperative copies).
// ---------------------------------------------------------------------------
__device__ __forceinline__ void gemm_tile_step(const bf16* a, const bf16* b,
                                               v8f* acc, int wave, int lm,
                                               int hi8, int khi)
{
    v16bf fa = load_frag16(&a[(wave * 16 + lm) * 32 + hi8],
                           &a[(wave * 16 + lm) * 32 + 16 + hi8]);
#pragma unroll
    for (int nt = 0; nt < 8; ++nt) {
        v16bf fb = load_frag16(&b[(nt * 16 + lm) * 32 + hi8],
                               &b[(nt * 16 + lm) * 32 + 16 + hi8]);
        acc[nt] = wmma_bf(fa, fb, acc[nt]);
    }
}

__global__ __launch_bounds__(256)
void gemm_bf16_n128(const bf16* __restrict__ A, int lda,
                    const bf16* __restrict__ Bt, int ldb,
                    float* __restrict__ C, int kchunks)
{
    __shared__ __align__(16) bf16 sA[2][128 * 32];
    __shared__ __align__(16) bf16 sB[2][128 * 32];
    const int t = threadIdx.x, lane = t & 31, wave = t >> 5;
    const int mbase = blockIdx.x * 128;
    const int hi8 = (lane >= 16) ? 8 : 0;
    const int khi = (lane >= 16) ? 16 : 0;
    const int lm  = lane & 15;

    v8f zero = {};
    v8f acc[8];
#pragma unroll
    for (int i = 0; i < 8; ++i) acc[i] = zero;

#if USE_TDM
    if (t == 0) {
        tdm_load_tile_2d(&sA[0][0], A + (size_t)mbase * lda, 32, 128, (unsigned long long)lda);
        tdm_load_tile_2d(&sB[0][0], Bt, 32, 128, (unsigned long long)ldb);
    }
    for (int kc = 0; kc < kchunks; ++kc) {
        int cur = kc & 1;
        if (t == 0) {
            if (kc + 1 < kchunks) {
                tdm_load_tile_2d(&sA[cur ^ 1][0],
                                 A + (size_t)mbase * lda + (size_t)(kc + 1) * 32,
                                 32, 128, (unsigned long long)lda);
                tdm_load_tile_2d(&sB[cur ^ 1][0], Bt + (size_t)(kc + 1) * 32,
                                 32, 128, (unsigned long long)ldb);
                tdm_wait(2);   // current pair complete, next pair in flight
            } else {
                tdm_wait(0);
            }
        }
        __syncthreads();
        gemm_tile_step(sA[cur], sB[cur], acc, wave, lm, hi8, khi);
        __syncthreads();
    }
#else
    for (int kc = 0; kc < kchunks; ++kc) {
        int row = t >> 1, koff = (t & 1) * 16;
        {
            const bf16* src = A + (size_t)(mbase + row) * lda + kc * 32 + koff;
            uint4* d = (uint4*)&sA[0][row * 32 + koff];
            d[0] = ((const uint4*)src)[0];
            d[1] = ((const uint4*)src)[1];
            if (kc + 1 < kchunks) __builtin_prefetch(src + 32, 0, 0);
        }
        {
            const bf16* src = Bt + (size_t)row * ldb + kc * 32 + koff;
            uint4* d = (uint4*)&sB[0][row * 32 + koff];
            d[0] = ((const uint4*)src)[0];
            d[1] = ((const uint4*)src)[1];
        }
        __syncthreads();
        gemm_tile_step(sA[0], sB[0], acc, wave, lm, hi8, khi);
        __syncthreads();
    }
#endif

#pragma unroll
    for (int nt = 0; nt < 8; ++nt)
#pragma unroll
        for (int r = 0; r < 8; ++r) {
            int row = mbase + wave * 16 + r + hi8;
            C[(size_t)row * 128 + nt * 16 + lm] = acc[nt][r];
        }
}

// ---------------------------------------------------------------------------
// fc1: x_new[16][256] += g[16][128000] * W[128000][256], split-K WMMA,
// on-the-fly f32->bf16 of W, f32 atomic reduction.
// ---------------------------------------------------------------------------
__global__ __launch_bounds__(256)
void fc1_splitk(const bf16* __restrict__ g, const float* __restrict__ w,
                float* __restrict__ xnew, int chunks_per)
{
    __shared__ __align__(16) bf16 sA[16 * 32];
    __shared__ __align__(16) bf16 sBt[256 * 32];
    const int t = threadIdx.x, lane = t & 31, wave = t >> 5;
    const int hi8 = (lane >= 16) ? 8 : 0;
    const int khi = (lane >= 16) ? 16 : 0;
    const int lm  = lane & 15;
    const int kc0 = blockIdx.x * chunks_per;

    v8f zero = {};
    v8f acc[2];
    acc[0] = zero; acc[1] = zero;

    for (int kk = 0; kk < chunks_per; ++kk) {
        int k0 = (kc0 + kk) * 32;
        for (int i = t; i < 16 * 32; i += 256) {
            int b = i >> 5, k = i & 31;
            sA[i] = g[(size_t)b * 128000 + k0 + k];
        }
        for (int i = 0; i < 32; ++i) {
            int flat = t * 32 + i;
            int r = flat >> 8, n = flat & 255;
            sBt[n * 32 + r] = (bf16)w[(size_t)(k0 + r) * 256 + n];
        }
        __syncthreads();
        v16bf fa = load_frag16(&sA[lm * 32 + hi8], &sA[lm * 32 + 16 + hi8]);
#pragma unroll
        for (int j = 0; j < 2; ++j) {
            int nt = wave * 2 + j;
            v16bf fb = load_frag16(&sBt[(nt * 16 + lm) * 32 + hi8],
                                   &sBt[(nt * 16 + lm) * 32 + 16 + hi8]);
            acc[j] = wmma_bf(fa, fb, acc[j]);
        }
        __syncthreads();
    }
#pragma unroll
    for (int j = 0; j < 2; ++j) {
        int nt = wave * 2 + j;
#pragma unroll
        for (int r = 0; r < 8; ++r)
            atomicAdd(&xnew[(r + hi8) * 256 + nt * 16 + lm], acc[j][r]);
    }
}

// ----------------------------- small kernels -------------------------------
__global__ void zero_f32(float* p, int n) {
    int i = blockIdx.x * 256 + threadIdx.x;
    if (i < n) p[i] = 0.f;
}
__global__ void fill_tail_kernel(bf16* h2, const float* gf) {
    int s = blockIdx.x, t = threadIdx.x;  // 32 threads: cols 7680..7711
    bf16 v = (bf16)0.0f;
    if (t == 0) v = (bf16)gf[(size_t)s * 2];
    else if (t == 1) v = (bf16)gf[(size_t)s * 2 + 1];
    h2[(size_t)s * KPAD + 7680 + t] = v;
}
// transposed bf16 weights: W1t[n][k] (n<128, k<KPAD), zero K-pad
__global__ void cvt_w1t(const float* w, bf16* o) {
    int i = blockIdx.x * 256 + threadIdx.x;
    if (i < 128 * KPAD) {
        int n = i / KPAD, k = i - n * KPAD;
        o[i] = (k < 7682) ? (bf16)w[(size_t)k * 128 + n] : (bf16)0.0f;
    }
}
__global__ void cvt_w2t(const float* w, bf16* o) {
    int i = blockIdx.x * 256 + threadIdx.x;
    if (i < 128 * 128) {
        int n = i >> 7, k = i & 127;
        o[i] = (bf16)w[k * 128 + n];
    }
}
__global__ void relu_cvt(const float* a, bf16* o, int n) {
    int i = blockIdx.x * 256 + threadIdx.x;
    if (i < n) o[i] = (bf16)fmaxf(a[i], 0.f);
}
__global__ void tanh_cvt(const float* a, bf16* o, int n) {
    int i = blockIdx.x * 256 + threadIdx.x;
    if (i < n) o[i] = (bf16)tanhf(a[i]);
}
__global__ void deg_accum(const int* ei, const float* ew, float* deg) {
    int e = blockIdx.x * 256 + threadIdx.x;
    if (e < NEDGE) atomicAdd(&deg[ei[NEDGE + e]], ew[e]);       // col = ei[1][e]
    else if (e < NEP) atomicAdd(&deg[e - NEDGE], 1.0f);         // self loops
}
__global__ void norm_build(const int* ei, const float* ew, const float* deg,
                           int* row, int* col, float* nrm) {
    int e = blockIdx.x * 256 + threadIdx.x;
    if (e >= NEP) return;
    int r, c; float w;
    if (e < NEDGE) { r = ei[e]; c = ei[NEDGE + e]; w = ew[e]; }
    else           { r = c = e - NEDGE; w = 1.0f; }
    float dr = deg[r], dc = deg[c];
    float ir = dr > 0.f ? rsqrtf(dr) : 0.f;
    float ic = dc > 0.f ? rsqrtf(dc) : 0.f;
    row[e] = r; col[e] = c; nrm[e] = ir * w * ic;
}
__global__ void aggregate(const float* __restrict__ m, const int* __restrict__ row,
                          const int* __restrict__ col, const float* __restrict__ nrm,
                          float* __restrict__ out) {
    int e = blockIdx.x, b = blockIdx.y, f = threadIdx.x;  // 128 threads = F
    int r = row[e], c = col[e];
    float w = nrm[e];
    float v = m[((size_t)b * N_NODES + r) * 128 + f] * w;
    atomicAdd(&out[((size_t)b * N_NODES + c) * 128 + f], v);
}
__global__ void init_xnew(const float* b, float* xnew) {
    int i = blockIdx.x * 256 + threadIdx.x;
    if (i < 16 * 256) xnew[i] = b[i & 255];
}
__global__ void fc2_kernel(const float* __restrict__ xnew, const float* __restrict__ w,
                           const float* __restrict__ b, float* __restrict__ out) {
    __shared__ float sx[256];
    int kb = blockIdx.x, k = kb >> 4, bb = kb & 15, t = threadIdx.x;
    sx[t] = xnew[bb * 256 + t];
    __syncthreads();
    for (int n = t; n < N_NODES; n += 256) {
        float s = b[k * N_NODES + n];
        for (int h = 0; h < 256; ++h)
            s += sx[h] * w[((size_t)k * 256 + h) * N_NODES + n];
        out[((size_t)k * 16 + bb) * N_NODES + n] = s;
    }
}

// ---------------------------------------------------------------------------
extern "C" void kernel_launch(void* const* d_in, const int* in_sizes, int n_in,
                              void* d_out, int out_size, void* d_ws, size_t ws_size,
                              hipStream_t stream)
{
    (void)in_sizes; (void)n_in; (void)out_size; (void)ws_size;
    const float* x   = (const float*)d_in[0];
    const int*   ei  = (const int*)d_in[1];
    const float* ew  = (const float*)d_in[2];
    const float* gf  = (const float*)d_in[3];
    const float* w1  = (const float*)d_in[4];
    const float* b1  = (const float*)d_in[5];
    const float* w2  = (const float*)d_in[6];
    const float* b2  = (const float*)d_in[7];
    const float* gw1 = (const float*)d_in[8];
    const float* gw2 = (const float*)d_in[9];
    const float* fw1 = (const float*)d_in[10];
    const float* fb1 = (const float*)d_in[11];
    const float* fw2 = (const float*)d_in[12];
    const float* fb2 = (const float*)d_in[13];
    float* out = (float*)d_out;

    char* ws = (char*)d_ws;
    size_t off = 0;
    auto alloc = [&](size_t bytes) {
        char* p = ws + off;
        off = (off + bytes + 255) & ~(size_t)255;
        return p;
    };
    bf16*  h2   = (bf16*)alloc((size_t)NSAMP * KPAD * 2);   // 247 MB bf16 activations
    bf16*  W1t  = (bf16*)alloc((size_t)128 * KPAD * 2);     // transposed [n][k]
    bf16*  W2t  = (bf16*)alloc((size_t)128 * 128 * 2);
    float* m    = (float*)alloc((size_t)NSAMP * 128 * 4);   // pre-aggregation
    float* a    = (float*)alloc((size_t)NSAMP * 128 * 4);   // post-aggregation
    bf16*  h1b  = (bf16*)alloc((size_t)NSAMP * 128 * 2);
    bf16*  h3   = (bf16*)alloc((size_t)NSAMP * 128 * 2);
    float* deg  = (float*)alloc(N_NODES * 4);
    int*   row  = (int*)alloc(NEP * 4);
    int*   col  = (int*)alloc(NEP * 4);
    float* nrm  = (float*)alloc(NEP * 4);
    float* xnew = (float*)alloc(16 * 256 * 4);

    const int NF = NSAMP * 128;  // 2,048,000

    conv_fused_kernel<<<NSAMP / S_PER_WG, 256, 0, stream>>>(x, w1, b1, w2, b2, h2);
    fill_tail_kernel<<<NSAMP, 32, 0, stream>>>(h2, gf);
    cvt_w1t<<<(128 * KPAD + 255) / 256, 256, 0, stream>>>(gw1, W1t);
    cvt_w2t<<<(128 * 128 + 255) / 256, 256, 0, stream>>>(gw2, W2t);

    zero_f32<<<(N_NODES + 255) / 256, 256, 0, stream>>>(deg, N_NODES);
    deg_accum<<<(NEP + 255) / 256, 256, 0, stream>>>(ei, ew, deg);
    norm_build<<<(NEP + 255) / 256, 256, 0, stream>>>(ei, ew, deg, row, col, nrm);

    // GCN1: [16000 x 7712] x [7712 x 128]
    gemm_bf16_n128<<<NSAMP / 128, 256, 0, stream>>>(h2, KPAD, W1t, KPAD, m, KPAD / 32);
    zero_f32<<<(NF + 255) / 256, 256, 0, stream>>>(a, NF);
    dim3 ag(NEP, B_SZ);
    aggregate<<<ag, 128, 0, stream>>>(m, row, col, nrm, a);
    relu_cvt<<<(NF + 255) / 256, 256, 0, stream>>>(a, h1b, NF);

    // GCN2: [16000 x 128] x [128 x 128]
    gemm_bf16_n128<<<NSAMP / 128, 256, 0, stream>>>(h1b, 128, W2t, 128, m, 4);
    zero_f32<<<(NF + 255) / 256, 256, 0, stream>>>(a, NF);
    aggregate<<<ag, 128, 0, stream>>>(m, row, col, nrm, a);
    tanh_cvt<<<(NF + 255) / 256, 256, 0, stream>>>(a, h3, NF);

    // fc1: [16 x 128000] x [128000 x 256] split-K (80 WGs x 50 chunks = 4000)
    init_xnew<<<16, 256, 0, stream>>>(fb1, xnew);
    fc1_splitk<<<80, 256, 0, stream>>>(h3, fw1, xnew, 50);

    // fc2: einsum('bh,khn->kbn') + bias
    fc2_kernel<<<80, 256, 0, stream>>>(xnew, fw2, fb2, out);
}
